// GraphClassifier_89953795047631
// MI455X (gfx1250) — matrix operand order repaired and checked by reading, not compiled, not run on hardware
//
#include <hip/hip_runtime.h>

typedef __attribute__((ext_vector_type(16))) __bf16 v16bf;
typedef __attribute__((ext_vector_type(8)))  __bf16 bf16x8;
typedef __attribute__((ext_vector_type(4)))  __bf16 bf16x4;
typedef __attribute__((ext_vector_type(8)))  float  v8f;

#define BM 64
#define BN 128
#define BK 32
#define LDSK 48   // LDS row stride in bf16 elements (96 B: 16B-aligned chunks, 32B row base)

// ---------------- degree / normalization kernels ----------------

__global__ __launch_bounds__(256) void k_init_deg(float* deg, int n) {
    int i = blockIdx.x * 256 + threadIdx.x;
    if (i < n) deg[i] = 1.0f;  // self-loop
}

__global__ __launch_bounds__(256) void k_accum_deg(const int* __restrict__ col,
                                                   float* deg, int e) {
    int i = blockIdx.x * 256 + threadIdx.x;
    if (i < e) unsafeAtomicAdd(&deg[col[i]], 1.0f);
}

__global__ __launch_bounds__(256) void k_rsqrt(float* deg, int n) {
    int i = blockIdx.x * 256 + threadIdx.x;
    if (i < n) deg[i] = rsqrtf(deg[i]);  // deg >= 1 always
}

// agg[i,f] = xw[i,f] * dinv[i]^2   (self-loop contribution, also zero-inits agg)
__global__ __launch_bounds__(256) void k_agg_init(const float* __restrict__ xw,
                                                  const float* __restrict__ dinv,
                                                  float* __restrict__ agg, int n) {
    long i = (long)blockIdx.x * 256 + threadIdx.x;
    long total = (long)n * 128;
    if (i < total) {
        int node = (int)(i >> 7);
        float d = dinv[node];
        agg[i] = xw[i] * d * d;
    }
}

// one wave of 32 lanes per edge; each lane handles 4 of the 128 features
__global__ __launch_bounds__(256) void k_agg_edges(const int* __restrict__ row,
                                                   const int* __restrict__ colv,
                                                   const float* __restrict__ dinv,
                                                   const float* __restrict__ xw,
                                                   float* __restrict__ agg, int e) {
    long t = (long)blockIdx.x * 256 + threadIdx.x;
    int edge = (int)(t >> 5);
    int lane = (int)(t & 31);
    if (edge < e) {
        int r = row[edge], c = colv[edge];
        float w = dinv[r] * dinv[c];
        const float* src = xw + (long)r * 128;
        float* dst = agg + (long)c * 128;
#pragma unroll
        for (int f = 0; f < 128; f += 32)
            unsafeAtomicAdd(&dst[f + lane], src[f + lane] * w);
    }
}

// BT[n][k] = bf16(B[k][n]) for n < N, else 0   (transposed, padded to Npad rows)
__global__ __launch_bounds__(256) void k_prep_bt(const float* __restrict__ B,
                                                 __bf16* __restrict__ BT,
                                                 int K, int N, int Npad) {
    long i = (long)blockIdx.x * 256 + threadIdx.x;
    long total = (long)Npad * K;
    if (i < total) {
        int n = (int)(i / K);
        int k = (int)(i - (long)n * K);
        float v = (n < N) ? B[(size_t)k * N + n] : 0.0f;
        BT[i] = (__bf16)v;
    }
}

// ---------------- fused bf16-WMMA GEMM ----------------
// C[M x Ncols] = act( concat(A0[M x K0], relu(A1[M x (K-K0)] + bA1)) * B + bias )
// BT is the pre-transposed bf16 weight [Npad][K], Npad multiple of BN, pad rows zeroed.
// Requires: K, K0 multiples of BK.
__global__ __launch_bounds__(256)
void k_gemm_bf16(const float* __restrict__ A0, int K0,
                 const float* __restrict__ A1, const float* __restrict__ bA1,
                 const __bf16* __restrict__ BT,
                 const float* __restrict__ bias,
                 float* __restrict__ C,
                 int M, int Ncols, int K, int relu_out) {
    __shared__ __attribute__((aligned(32))) __bf16 sA[BM][LDSK];   // [row][k]
    __shared__ __attribute__((aligned(32))) __bf16 sBT[BN][LDSK];  // [col][k]

    const int tid  = threadIdx.x;
    const int wave = tid >> 5;
    const int lane = tid & 31;
    const int wm   = wave & 1;   // 0..1 : 32-row band
    const int wn   = wave >> 1;  // 0..3 : 32-col band
    const int m16  = lane & 15;
    const int half = lane >> 4;

    const int blockM = blockIdx.y * BM;
    const int blockN = blockIdx.x * BN;
    const int K1 = K - K0;

    v8f acc[2][2];
#pragma unroll
    for (int i = 0; i < 2; ++i)
#pragma unroll
        for (int j = 0; j < 2; ++j) acc[i][j] = (v8f){0,0,0,0,0,0,0,0};

    for (int k0 = 0; k0 < K; k0 += BK) {
        // ---- B tile: pure bf16 copy global->LDS via async DMA (ASYNCcnt path) ----
        // BN rows x 64B; 16B per lane-transfer; 2 async instructions per wave.
#pragma unroll
        for (int it = 0; it < 2; ++it) {
            int q  = tid + it * 256;       // 0..511
            int nn = q >> 2;               // row 0..127
            int kc = (q & 3) << 3;         // bf16 element offset: 0,8,16,24
            const __bf16* gp = BT + (size_t)(blockN + nn) * K + (k0 + kc);
            unsigned ldst = (unsigned)(uintptr_t)(&sBT[nn][kc]);  // low 32 bits = LDS offset
            asm volatile("global_load_async_to_lds_b128 %0, %1, off"
                         :: "v"(ldst), "v"(gp)
                         : "memory");
        }

        // ---- A tile: BM x BK, float4 loads along K, cvt to bf16, LDS stores ----
        // K0 % BK == 0  =>  the concat branch is uniform per K-tile.
#pragma unroll
        for (int it = 0; it < (BM * BK / 4) / 256; ++it) {
            int q  = tid + it * 256;
            int r  = q >> 3;          // 8 quads per row
            int kq = (q & 7) << 2;
            int gr = blockM + r;
            float4 v = float4{0.f, 0.f, 0.f, 0.f};
            if (k0 < K0) {
                if (gr < M)
                    v = *(const float4*)(A0 + (size_t)gr * K0 + (k0 + kq));
            } else {
                int kr = k0 - K0 + kq;
                if (gr < M) {
                    float4 t = *(const float4*)(A1 + (size_t)gr * K1 + kr);
                    v.x = fmaxf(t.x + bA1[kr + 0], 0.f);
                    v.y = fmaxf(t.y + bA1[kr + 1], 0.f);
                    v.z = fmaxf(t.z + bA1[kr + 2], 0.f);
                    v.w = fmaxf(t.w + bA1[kr + 3], 0.f);
                }
            }
            bf16x4 p;
            p[0] = (__bf16)v.x; p[1] = (__bf16)v.y;
            p[2] = (__bf16)v.z; p[3] = (__bf16)v.w;
            *(bf16x4*)&sA[r][kq] = p;
        }

        asm volatile("s_wait_asynccnt 0x0" ::: "memory");
        __syncthreads();

        // ---- per-lane fragments: all contiguous 16B ds_load_b128 runs ----
        v16bf aF[2], bF[2];
#pragma unroll
        for (int i = 0; i < 2; ++i) {
            const __bf16* pr = &sA[wm * 32 + i * 16 + m16][0];
            bf16x8 lo = *(const bf16x8*)(pr + half * 8);        // k in half*8 + [0,8)
            bf16x8 hi = *(const bf16x8*)(pr + 16 + half * 8);   // k in 16+half*8 + [0,8)
            aF[i] = __builtin_shufflevector(lo, hi, 0,1,2,3,4,5,6,7,8,9,10,11,12,13,14,15);
        }
#pragma unroll
        for (int j = 0; j < 2; ++j) {
            const __bf16* pc = &sBT[wn * 32 + j * 16 + m16][0];  // k in half*8 + [0,8) and 16+half*8 + [0,8)
            bf16x8 lo = *(const bf16x8*)(pc + half * 8);
            bf16x8 hi = *(const bf16x8*)(pc + 16 + half * 8);
            bF[j] = __builtin_shufflevector(lo, hi, 0,1,2,3,4,5,6,7,8,9,10,11,12,13,14,15);
        }

#pragma unroll
        for (int i = 0; i < 2; ++i)
#pragma unroll
            for (int j = 0; j < 2; ++j)
                acc[i][j] = __builtin_amdgcn_wmma_f32_16x16x32_bf16(
                    false, aF[i], false, bF[j], (short)0, acc[i][j], false, false);

        __syncthreads();
    }

    // ---- epilogue: bias + optional relu, masked store ----
#pragma unroll
    for (int i = 0; i < 2; ++i) {
#pragma unroll
        for (int j = 0; j < 2; ++j) {
            int col = blockN + wn * 32 + j * 16 + m16;
            float bv = (bias && col < Ncols) ? bias[col] : 0.0f;
#pragma unroll
            for (int r = 0; r < 8; ++r) {
                int rowO = blockM + wm * 32 + i * 16 + half * 8 + r;
                if (rowO < M && col < Ncols) {
                    float v = acc[i][j][r] + bv;
                    if (relu_out) v = v > 0.0f ? v : 0.0f;
                    C[(size_t)rowO * Ncols + col] = v;
                }
            }
        }
    }
}

// ---------------- launcher ----------------

extern "C" void kernel_launch(void* const* d_in, const int* in_sizes, int n_in,
                              void* d_out, int out_size, void* d_ws, size_t ws_size,
                              hipStream_t stream) {
    const float* x   = (const float*)d_in[0];
    const int*   ei  = (const int*)d_in[1];   // edge_index [2, E]
    // d_in[2] img_sizes: unused by reference math
    const float* Wg  = (const float*)d_in[3]; // [1280,128]
    const float* bg  = (const float*)d_in[4]; // [128]
    const float* Wh  = (const float*)d_in[5]; // [1408,512]
    const float* bh  = (const float*)d_in[6]; // [512]
    const float* Wc  = (const float*)d_in[7]; // [512,79]
    const float* bc  = (const float*)d_in[8]; // [79]

    const int IN_CH = 1280, G_CH = 128, HID = 512, NCLS = 79;
    const int KCAT = IN_CH + G_CH;       // 1408
    const int NCLS_PAD = 128;
    const int n = in_sizes[0] / IN_CH;
    const int e = in_sizes[1] / 2;
    const int* row = ei;
    const int* col = ei + e;

    // ---- workspace carve-out (256B aligned chunks) ----
    char* wsb = (char*)d_ws;
    size_t off = 0;
    auto carve = [&](size_t bytes) {
        char* p = wsb + off;
        off += (bytes + 255) & ~(size_t)255;
        return p;
    };
    float*  deg  = (float*)carve((size_t)n * sizeof(float));
    float*  xw   = (float*)carve((size_t)n * G_CH * sizeof(float));
    float*  agg  = (float*)carve((size_t)n * G_CH * sizeof(float));
    __bf16* bt_g = (__bf16*)carve((size_t)G_CH * IN_CH * sizeof(__bf16));      // 128x1280
    __bf16* bt_h = (__bf16*)carve((size_t)HID * KCAT * sizeof(__bf16));        // 512x1408
    __bf16* bt_c = (__bf16*)carve((size_t)NCLS_PAD * HID * sizeof(__bf16));    // 128x512
    (void)ws_size;

    float* h    = (float*)d_out;                       // [n,512]
    float* logi = h + (size_t)n * HID;                 // [n,79]

    dim3 blk(256);

    // weight prep: bf16, transposed, padded
    {
        long t1 = (long)G_CH * IN_CH;
        long t2 = (long)HID * KCAT;
        long t3 = (long)NCLS_PAD * HID;
        k_prep_bt<<<dim3((unsigned)((t1 + 255) / 256)), blk, 0, stream>>>(Wg, bt_g, IN_CH, G_CH, G_CH);
        k_prep_bt<<<dim3((unsigned)((t2 + 255) / 256)), blk, 0, stream>>>(Wh, bt_h, KCAT, HID, HID);
        k_prep_bt<<<dim3((unsigned)((t3 + 255) / 256)), blk, 0, stream>>>(Wc, bt_c, HID, NCLS, NCLS_PAD);
    }

    // degrees -> dinv
    k_init_deg<<<dim3((n + 255) / 256), blk, 0, stream>>>(deg, n);
    k_accum_deg<<<dim3((e + 255) / 256), blk, 0, stream>>>(col, deg, e);
    k_rsqrt<<<dim3((n + 255) / 256), blk, 0, stream>>>(deg, n);

    // GEMM1: xw = x @ W_gcn   (no bias, no relu)
    k_gemm_bf16<<<dim3(G_CH / BN, (n + BM - 1) / BM), blk, 0, stream>>>(
        x, IN_CH, nullptr, nullptr, bt_g, nullptr, xw, n, G_CH, IN_CH, 0);

    // aggregation: self-loop init + edge scatter-add
    {
        long total = (long)n * G_CH;
        k_agg_init<<<dim3((unsigned)((total + 255) / 256)), blk, 0, stream>>>(xw, deg, agg, n);
        long et = (long)e * 32;
        k_agg_edges<<<dim3((unsigned)((et + 255) / 256)), blk, 0, stream>>>(row, col, deg, xw, agg, e);
    }

    // GEMM2: h = relu(concat(x, relu(agg + b_gcn)) @ W_hid + b_hid)
    k_gemm_bf16<<<dim3(HID / BN, (n + BM - 1) / BM), blk, 0, stream>>>(
        x, IN_CH, agg, bg, bt_h, bh, h, n, HID, KCAT, 1);

    // GEMM3: logits = h @ W_cls + b_cls
    k_gemm_bf16<<<dim3(NCLS_PAD / BN, (n + BM - 1) / BM), blk, 0, stream>>>(
        h, HID, nullptr, nullptr, bt_c, bc, logi, n, NCLS, HID, 0);
}
